// ContextAttentionBlock_747324310309
// MI455X (gfx1250) — compile-verified, matrix operand order checked by reading, not run on hardware
//
#include <hip/hip_runtime.h>
#include <hip/hip_bf16.h>

// ---------------------------------------------------------------------------
// ContextAttentionBlock for MI455X (gfx1250, wave32, WMMA f16->f32, TDM)
// B=4, C=256, H=W=64 -> N=4096, CQK=32, HIDDEN=100
// ---------------------------------------------------------------------------

#define BB   4
#define CC   256
#define NN   4096
#define DQK  32
#define HID  100
#define JC   128            // attention j-chunk per outer iteration
#define PJ   (JC + 8)       // Pshared row stride in halves (16B-aligned rows)

typedef __attribute__((ext_vector_type(16))) _Float16 v16h;
typedef __attribute__((ext_vector_type(8)))  _Float16 v8h;
typedef __attribute__((ext_vector_type(8)))  float    v8f;
typedef __attribute__((ext_vector_type(4)))  unsigned int u32x4;
typedef __attribute__((ext_vector_type(8)))  int          i32x8;
typedef __attribute__((ext_vector_type(4)))  int          i32x4;

#if __has_builtin(__builtin_amdgcn_tensor_load_to_lds)
#define HAVE_TDM 1
#else
#define HAVE_TDM 0
#endif

static __device__ __forceinline__ v16h cat8(v8h lo, v8h hi) {
    v16h r;
#pragma unroll
    for (int i = 0; i < 8; ++i) { r[i] = lo[i]; r[i + 8] = hi[i]; }
    return r;
}

static __device__ __forceinline__ v8f wmma_f16(v16h a, v16h b, v8f c) {
    return __builtin_amdgcn_wmma_f32_16x16x32_f16(false, a, false, b,
                                                  (short)0, c, false, false);
}

#if HAVE_TDM
// 2D TDM descriptor: tile [CC rows x JC cols] of 2-byte elements out of a
// row-major [CC x NN] tensor (row stride NN), DMA'd Global -> LDS.
// Field packing per CDNA5 ISA.
static __device__ __forceinline__ void tdm_load_v_tile(const _Float16* gsrc,
                                                       unsigned lds_off) {
    const unsigned long long ga = (unsigned long long)(uintptr_t)gsrc;
    u32x4 g0;
    g0[0] = 1u;                                    // count=1, user descriptor
    g0[1] = lds_off;                               // lds_addr
    g0[2] = (unsigned)(ga & 0xffffffffu);          // global_addr[31:0]
    g0[3] = (unsigned)((ga >> 32) & 0x01ffffffu)   // global_addr[56:32]
          | 0x80000000u;                           // type = 2
    i32x8 g1;
    g1[0] = 0x00010000;                    // wg_mask=0, data_size=1 (2 bytes)
    g1[1] = (int)((NN & 0xffffu) << 16);   // tensor_dim0 = NN   (bits 79:48)
    g1[2] = (int)((CC & 0xffffu) << 16);   // tensor_dim1 = CC   (bits 111:80)
    g1[3] = (int)(JC << 16);               // tile_dim0  = JC    (bits 127:112)
    g1[4] = (int)CC;                       // tile_dim1  = CC, tile_dim2 = 0
    g1[5] = (int)NN;                       // tensor_dim0_stride = NN
    g1[6] = 0;
    g1[7] = 0;
    i32x4 g2 = {0, 0, 0, 0};
    i32x4 g3 = {0, 0, 0, 0};
#if defined(__clang_major__) && (__clang_major__ >= 23)
    i32x8 g4 = {0, 0, 0, 0, 0, 0, 0, 0};
    __builtin_amdgcn_tensor_load_to_lds(g0, g1, g2, g3, g4, 0);
#else
    __builtin_amdgcn_tensor_load_to_lds(g0, g1, g2, g3, 0);
#endif
}
#endif

// ---------------------------------------------------------------------------
// Kernel 1: q/k/v projections (unchanged).
//   qT/kT: [B][N][32] f16 ; vh: [B][256][N] f16
// ---------------------------------------------------------------------------
__global__ __launch_bounds__(128) void qkv_kernel(
    const float* __restrict__ xf,
    const float* __restrict__ Wq, const float* __restrict__ bq,
    const float* __restrict__ Wk, const float* __restrict__ bk,
    const float* __restrict__ Wv, const float* __restrict__ bv,
    _Float16* __restrict__ qT, _Float16* __restrict__ kT,
    _Float16* __restrict__ vh)
{
    const int b  = blockIdx.z;
    const int rt = blockIdx.y;
    const int nb = blockIdx.x * 64;

    __shared__ alignas(16) _Float16 Wt[16][264];
    __shared__ alignas(16) _Float16 Xt[64][264];

    const float* Wsel; const float* bsel; int row0, mode;
    if      (rt < 2) { Wsel = Wq; bsel = bq; row0 = rt * 16;       mode = 0; }
    else if (rt < 4) { Wsel = Wk; bsel = bk; row0 = (rt - 2) * 16; mode = 1; }
    else             { Wsel = Wv; bsel = bv; row0 = (rt - 4) * 16; mode = 2; }

    for (int t = threadIdx.x; t < 16 * 256; t += 128) {
        const int m = t >> 8, c = t & 255;
        Wt[m][c] = (_Float16)Wsel[(row0 + m) * CC + c];
    }
    for (int t = threadIdx.x; t < 64 * 256; t += 128) {
        const int c = t >> 6, n = t & 63;
        Xt[n][c] = (_Float16)xf[((size_t)b * CC + c) * NN + nb + n];
    }
    __syncthreads();

    const int lane = threadIdx.x & 31;
    const int wv   = threadIdx.x >> 5;
    const int li   = lane & 15;
    const int hh   = lane >> 4;
    const int nloc = wv * 16 + li;

    v8f acc = {};
#pragma unroll
    for (int kk = 0; kk < 256; kk += 32) {
        v16h af = cat8(*(const v8h*)&Wt[li][kk + 8 * hh],
                       *(const v8h*)&Wt[li][kk + 16 + 8 * hh]);
        v16h bf = cat8(*(const v8h*)&Xt[nloc][kk + 8 * hh],
                       *(const v8h*)&Xt[nloc][kk + 16 + 8 * hh]);
        acc = wmma_f16(af, bf, acc);
    }
#pragma unroll
    for (int r = 0; r < 8; ++r) acc[r] += bsel[row0 + r + 8 * hh];

    if (mode == 2) {
#pragma unroll
        for (int r = 0; r < 8; ++r) {
            const int c = row0 + r + 8 * hh;
            vh[((size_t)b * CC + c) * NN + nb + nloc] = (_Float16)acc[r];
        }
    } else {
        v8h out;
#pragma unroll
        for (int r = 0; r < 8; ++r) out[r] = (_Float16)acc[r];
        _Float16* dst = (mode == 0) ? qT : kT;
        *(v8h*)&dst[((size_t)b * NN + nb + nloc) * DQK + row0 + 8 * hh] = out;
    }
}

// ---------------------------------------------------------------------------
// Kernel 2: fused attention, v2.
//  - Scores computed once per block: wave w owns j sub-tile jb+16w of each
//    128-wide chunk; P^T shared through LDS (8x less score WMMA than v1).
//  - V tile [256 x 128] staged Global->LDS by the Tensor Data Mover, double
//    buffered, gated by s_wait_tensorcnt; WMMA A-operands become ds_load_b128.
// grid = (N/16, B), 256 thr = 8 waves; wave w owns channels [32w, 32w+32).
// ---------------------------------------------------------------------------
__global__ __launch_bounds__(256) void attn_kernel(
    const float* __restrict__ xf,
    const _Float16* __restrict__ qT, const _Float16* __restrict__ kT,
    const _Float16* __restrict__ vh, const float* __restrict__ gammap,
    float* __restrict__ attn)
{
    const int b  = blockIdx.y;
    const int qb = blockIdx.x * 16;
    const int wv   = threadIdx.x >> 5;
    const int lane = threadIdx.x & 31;
    const int li   = lane & 15;
    const int hh   = lane >> 4;
    const size_t bN = (size_t)b * NN;

    __shared__ alignas(16) _Float16 Pshared[16 * PJ];   // [i][j-in-chunk]
#if HAVE_TDM
    __shared__ alignas(128) _Float16 Vlds[2][CC * JC];  // double-buffered V
#endif
    __shared__ float lmb[8][16];
    __shared__ float lsb[8][16];

    const _Float16* vbase = vh + (size_t)b * CC * NN;

#if HAVE_TDM
    if (wv == 0)        // prefetch chunk 0; overlaps all of pass 1
        tdm_load_v_tile(vbase, (unsigned)(uintptr_t)&Vlds[0][0]);
#endif

    // Q fragment (A operand, 16 queries x 32 dims), loaded once.
    const _Float16* qp = qT + (bN + qb + li) * DQK;
    const v16h qfrag = cat8(*(const v8h*)(qp + 8 * hh),
                            *(const v8h*)(qp + 16 + 8 * hh));

    // ---- pass 1: row max, j space partitioned across the 8 waves ----
    float lm[8];
#pragma unroll
    for (int r = 0; r < 8; ++r) lm[r] = -3.4e38f;

    for (int jb = 16 * wv; jb < NN; jb += JC) {
        const _Float16* kp = kT + (bN + jb + li) * DQK;
        __builtin_prefetch(kp + JC * DQK, 0, 0);
        v16h kf = cat8(*(const v8h*)(kp + 8 * hh),
                       *(const v8h*)(kp + 16 + 8 * hh));
        v8f s = {};
        s = wmma_f16(qfrag, kf, s);
#pragma unroll
        for (int r = 0; r < 8; ++r) lm[r] = fmaxf(lm[r], s[r]);
    }
#pragma unroll
    for (int r = 0; r < 8; ++r)
        for (int off = 1; off < 16; off <<= 1)
            lm[r] = fmaxf(lm[r], __shfl_xor(lm[r], off));
    if (li == 0) {
#pragma unroll
        for (int r = 0; r < 8; ++r) lmb[wv][r + 8 * hh] = lm[r];
    }
    __syncthreads();
    float rm[8];
#pragma unroll
    for (int r = 0; r < 8; ++r) {
        float m = lmb[0][r + 8 * hh];
#pragma unroll
        for (int w = 1; w < 8; ++w) m = fmaxf(m, lmb[w][r + 8 * hh]);
        rm[r] = m;
    }

    // ---- pass 2: exp + shared P^T + O accumulation ----
    const int c0 = wv * 32;
    v8f O0 = {}, O1 = {};
    float ls[8];
#pragma unroll
    for (int r = 0; r < 8; ++r) ls[r] = 0.0f;

    for (int it = 0; it < NN / JC; ++it) {
        const int jb = it * JC;

        // scores for this wave's 16-wide j sub-tile
        {
            const int j16 = jb + 16 * wv;
            const _Float16* kp = kT + (bN + j16 + li) * DQK;
            v16h kf = cat8(*(const v8h*)(kp + 8 * hh),
                           *(const v8h*)(kp + 16 + 8 * hh));
            v8f s = {};
            s = wmma_f16(qfrag, kf, s);
#pragma unroll
            for (int r = 0; r < 8; ++r) {
                const float p = __expf(s[r] - rm[r]);
                ls[r] += p;
                Pshared[(r + 8 * hh) * PJ + 16 * wv + li] = (_Float16)p;
            }
        }
#if HAVE_TDM
        if (wv == 0) __builtin_amdgcn_s_wait_tensorcnt(0);  // chunk `it` landed
#endif
        __syncthreads();   // P^T published; V buffer `it&1` ready; old buf free
#if HAVE_TDM
        if (wv == 0 && it + 1 < NN / JC)
            tdm_load_v_tile(vbase + (size_t)(it + 1) * JC,
                            (unsigned)(uintptr_t)&Vlds[(it + 1) & 1][0]);
        const _Float16* Vcur = &Vlds[it & 1][0];
#endif
#pragma unroll
        for (int js = 0; js < JC / 32; ++js) {
            const int jo = js * 32;
            const _Float16* pb = &Pshared[li * PJ + jo];
            v16h pf = cat8(*(const v8h*)(pb + 8 * hh),
                           *(const v8h*)(pb + 16 + 8 * hh));
#if HAVE_TDM
            const _Float16* vp0 = Vcur + (c0 + li) * JC + jo;
            const _Float16* vp1 = Vcur + (c0 + 16 + li) * JC + jo;
#else
            const _Float16* vp0 = vbase + (size_t)(c0 + li) * NN + jb + jo;
            const _Float16* vp1 = vbase + (size_t)(c0 + 16 + li) * NN + jb + jo;
#endif
            v16h vf0 = cat8(*(const v8h*)(vp0 + 8 * hh),
                            *(const v8h*)(vp0 + 16 + 8 * hh));
            O0 = wmma_f16(vf0, pf, O0);
            v16h vf1 = cat8(*(const v8h*)(vp1 + 8 * hh),
                            *(const v8h*)(vp1 + 16 + 8 * hh));
            O1 = wmma_f16(vf1, pf, O1);
        }
        __syncthreads();   // everyone done with Pshared + Vcur
    }

    // row-sum: shfl-reduce within wave, then cross-wave sum through LDS
#pragma unroll
    for (int r = 0; r < 8; ++r)
        for (int off = 1; off < 16; off <<= 1)
            ls[r] += __shfl_xor(ls[r], off);
    if (li == 0) {
#pragma unroll
        for (int r = 0; r < 8; ++r) lsb[wv][r + 8 * hh] = ls[r];
    }
    __syncthreads();
    float rs = 0.0f;
#pragma unroll
    for (int w = 0; w < 8; ++w) rs += lsb[w][li];
    const float rinv  = 1.0f / rs;
    const float gamma = gammap[0];

#pragma unroll
    for (int r = 0; r < 8; ++r) {
        {
            const int c = c0 + r + 8 * hh;
            const size_t idx = ((size_t)b * CC + c) * NN + qb + li;
            attn[idx] = gamma * (O0[r] * rinv) + xf[idx];
        }
        {
            const int c = c0 + 16 + r + 8 * hh;
            const size_t idx = ((size_t)b * CC + c) * NN + qb + li;
            attn[idx] = gamma * (O1[r] * rinv) + xf[idx];
        }
    }
}

// ---------------------------------------------------------------------------
// Kernel 3: s[b,n] = sum_h ctx[h] * tanh( (Wfc @ attn[:, n])[h] + bfc[h] )
// ---------------------------------------------------------------------------
__global__ __launch_bounds__(256) void ctx_kernel(
    const float* __restrict__ attn, const float* __restrict__ Wfc,
    const float* __restrict__ bfc, const float* __restrict__ ctx,
    float* __restrict__ sarr)
{
    const int idx = blockIdx.x * 256 + threadIdx.x;
    const int b = idx >> 12, n = idx & (NN - 1);

    float acc[HID];
#pragma unroll
    for (int h = 0; h < HID; ++h) acc[h] = bfc[h];

    const float* ap = attn + (size_t)b * CC * NN + n;
    for (int c = 0; c < CC; ++c) {
        const float x = ap[(size_t)c * NN];
#pragma unroll
        for (int h = 0; h < HID; ++h) acc[h] += Wfc[h * CC + c] * x;
    }
    float s = 0.0f;
#pragma unroll
    for (int h = 0; h < HID; ++h) s += tanhf(acc[h]) * ctx[h];
    sarr[idx] = s;
}

// ---------------------------------------------------------------------------
// Kernel 4: a[b,:] = softmax_n(s[b,:])
// ---------------------------------------------------------------------------
__global__ __launch_bounds__(256) void pool_softmax_kernel(
    const float* __restrict__ sarr, float* __restrict__ abuf)
{
    const int b = blockIdx.x, tid = threadIdx.x;
    __shared__ float red[256];

    float m = -3.4e38f;
    for (int n = tid; n < NN; n += 256) m = fmaxf(m, sarr[b * NN + n]);
    red[tid] = m; __syncthreads();
    for (int s = 128; s > 0; s >>= 1) {
        if (tid < s) red[tid] = fmaxf(red[tid], red[tid + s]);
        __syncthreads();
    }
    const float mx = red[0]; __syncthreads();

    float sum = 0.0f;
    for (int n = tid; n < NN; n += 256) sum += __expf(sarr[b * NN + n] - mx);
    red[tid] = sum; __syncthreads();
    for (int s = 128; s > 0; s >>= 1) {
        if (tid < s) red[tid] += red[tid + s];
        __syncthreads();
    }
    const float inv = 1.0f / red[0];

    for (int n = tid; n < NN; n += 256)
        abuf[b * NN + n] = __expf(sarr[b * NN + n] - mx) * inv;
}

// ---------------------------------------------------------------------------
// Kernel 5: out[b,c] = sum_n xf[b,c,n] * a[b,n]
// ---------------------------------------------------------------------------
__global__ __launch_bounds__(256) void pool_out_kernel(
    const float* __restrict__ xf, const float* __restrict__ abuf,
    float* __restrict__ out)
{
    const int b = blockIdx.x, c = threadIdx.x;
    const float* xp = xf + ((size_t)b * CC + c) * NN;
    const float* ap = abuf + b * NN;
    float s = 0.0f;
    for (int n = 0; n < NN; ++n) s += xp[n] * ap[n];
    out[b * CC + c] = s;
}

// ---------------------------------------------------------------------------
extern "C" void kernel_launch(void* const* d_in, const int* in_sizes, int n_in,
                              void* d_out, int out_size, void* d_ws, size_t ws_size,
                              hipStream_t stream) {
    const float* xf    = (const float*)d_in[0];
    const float* Wq    = (const float*)d_in[1];
    const float* bq    = (const float*)d_in[2];
    const float* Wk    = (const float*)d_in[3];
    const float* bk    = (const float*)d_in[4];
    const float* Wv    = (const float*)d_in[5];
    const float* bv    = (const float*)d_in[6];
    const float* gamma = (const float*)d_in[7];
    const float* Wfc   = (const float*)d_in[8];
    const float* bfc   = (const float*)d_in[9];
    const float* ctx   = (const float*)d_in[10];
    float* out = (float*)d_out;

    char* ws = (char*)d_ws;
    _Float16* qT   = (_Float16*)(ws);
    _Float16* kT   = (_Float16*)(ws + (size_t)(1u  << 20));
    _Float16* vh   = (_Float16*)(ws + (size_t)(2u  << 20));
    float*    attn = (float*)   (ws + (size_t)(10u << 20));
    float*    sarr = (float*)   (ws + (size_t)(26u << 20));
    float*    abuf = (float*)   (ws + (size_t)(26u << 20) + (1u << 16));

    qkv_kernel<<<dim3(NN / 64, 20, BB), 128, 0, stream>>>(
        xf, Wq, bq, Wk, bk, Wv, bv, qT, kT, vh);

    attn_kernel<<<dim3(NN / 16, BB), 256, 0, stream>>>(
        xf, qT, kT, vh, gamma, attn);

    ctx_kernel<<<dim3((BB * NN) / 256), 256, 0, stream>>>(
        attn, Wfc, bfc, ctx, sarr);

    pool_softmax_kernel<<<dim3(BB), 256, 0, stream>>>(sarr, abuf);

    pool_out_kernel<<<dim3(BB), 256, 0, stream>>>(xf, abuf, out);
}
